// VisualEncoder_4629974745571
// MI455X (gfx1250) — matrix operand order repaired and checked by reading, not compiled, not run on hardware
//
#include <hip/hip_runtime.h>
#include <math.h>

// ---------------------------------------------------------------------------
// ViT visual encoder for MI455X (gfx1250, wave32).
// All large GEMMs -> v_wmma_f32_16x16x32_f16 (f32 accum). Tiles are staged in
// LDS pre-swizzled into WMMA fragment order so each lane fetches its whole
// 16-half fragment as one contiguous 32B read (ds_load_b128 pairs) instead of
// 16 scalar ds_load_u16 gathers. Compute-bound workload (~640 GFLOP vs
// ~150 MB of traffic @ 23.3 TB/s -> WMMA issue rate is the ceiling).
// ---------------------------------------------------------------------------

typedef __attribute__((ext_vector_type(16))) _Float16 v16h;
typedef __attribute__((ext_vector_type(4)))  _Float16 v4h;
typedef __attribute__((ext_vector_type(8)))  float    v8f;

#define BATCH 32
#define NTOK  257
#define DMODEL 512
#define NHEAD 8
#define HDIM 64
#define MLPD 2048
#define NLAYER 8
#define MROWS (BATCH * NTOK)      // 8224
#define EPSV 1e-5f

#define TILE_M 64
#define TILE_N 128
#define TILE_K 32

// ----------------------------- reductions ----------------------------------
__device__ __forceinline__ void block_sum2(float& s, float& ss) {
    __shared__ float sh_a[256];
    __shared__ float sh_b[256];
    int tid = threadIdx.x;
    sh_a[tid] = s; sh_b[tid] = ss;
    __syncthreads();
    for (int off = 128; off > 0; off >>= 1) {
        if (tid < off) { sh_a[tid] += sh_a[tid + off]; sh_b[tid] += sh_b[tid + off]; }
        __syncthreads();
    }
    s = sh_a[0]; ss = sh_b[0];
    __syncthreads();
}

__device__ __forceinline__ float block_max1(float v) {
    __shared__ float sh_m[256];
    int tid = threadIdx.x;
    sh_m[tid] = v;
    __syncthreads();
    for (int off = 128; off > 0; off >>= 1) {
        if (tid < off) sh_m[tid] = fmaxf(sh_m[tid], sh_m[tid + off]);
        __syncthreads();
    }
    float r = sh_m[0];
    __syncthreads();
    return r;
}

__device__ __forceinline__ float block_sum1(float v) {
    __shared__ float sh_s[256];
    int tid = threadIdx.x;
    sh_s[tid] = v;
    __syncthreads();
    for (int off = 128; off > 0; off >>= 1) {
        if (tid < off) sh_s[tid] += sh_s[tid + off];
        __syncthreads();
    }
    float r = sh_s[0];
    __syncthreads();
    return r;
}

// ----------------------------- conv stem -----------------------------------
__global__ void ve_conv3x3(const float* __restrict__ in, const float* __restrict__ w,
                           const float* __restrict__ bias, float* __restrict__ out,
                           int Cin, int Cout) {
    int idx = blockIdx.x * blockDim.x + threadIdx.x;
    int total = BATCH * Cout * 128 * 128;
    if (idx >= total) return;
    int px = idx & 127;
    int py = (idx >> 7) & 127;
    int rem = idx >> 14;
    int co = rem % Cout;
    int bn = rem / Cout;
    float s = bias[co];
    for (int ci = 0; ci < Cin; ++ci) {
        const float* ip = in + (size_t)(bn * Cin + ci) * 16384;
        const float* wp = w + (size_t)(co * Cin + ci) * 9;
        #pragma unroll
        for (int dy = -1; dy <= 1; ++dy) {
            int yy = py + dy;
            if (yy < 0 || yy > 127) continue;
            #pragma unroll
            for (int dx = -1; dx <= 1; ++dx) {
                int xx = px + dx;
                if (xx < 0 || xx > 127) continue;
                s += ip[yy * 128 + xx] * wp[(dy + 1) * 3 + (dx + 1)];
            }
        }
    }
    out[idx] = s;
}

__global__ __launch_bounds__(256) void ve_gn_stats(const float* __restrict__ x,
                                                   float* __restrict__ stats,
                                                   int C, int groups) {
    int bg = blockIdx.x;
    int g  = bg % groups;
    int bn = bg / groups;
    int cpg = C / groups;
    int n = cpg * 16384;
    const float* base = x + ((size_t)bn * C + (size_t)g * cpg) * 16384;
    float s = 0.f, ss = 0.f;
    for (int i = threadIdx.x; i < n; i += 256) {
        float v = base[i];
        s += v; ss += v * v;
    }
    block_sum2(s, ss);
    if (threadIdx.x == 0) {
        float m = s / (float)n;
        stats[bg * 2 + 0] = m;
        stats[bg * 2 + 1] = ss / (float)n - m * m;
    }
}

__global__ void ve_gn_apply_gelu(float* __restrict__ x, const float* __restrict__ stats,
                                 const float* __restrict__ gg, const float* __restrict__ gb,
                                 int C, int groups) {
    int idx = blockIdx.x * blockDim.x + threadIdx.x;
    int total = BATCH * C * 16384;
    if (idx >= total) return;
    int c  = (idx >> 14) % C;
    int bn = idx / (C * 16384);
    int g  = c / (C / groups);
    float m = stats[(bn * groups + g) * 2 + 0];
    float v = stats[(bn * groups + g) * 2 + 1];
    float t = (x[idx] - m) * rsqrtf(v + EPSV) * gg[c] + gb[c];
    x[idx] = 0.5f * t * (1.f + erff(t * 0.70710678118654752f));
}

// ----------------------------- WMMA GEMMs ----------------------------------
// Fragment-order LDS staging.
// A-layout inverse (ISA 7.12.2, 16-bit A 16x32): for tile coords (r, c):
//   lane = 16*((c>>3)&1) + (r&15), elem = (c&7) | ((c>>4)<<3), msub = r>>4.
// B-layout inverse (32x16): for (k, col):
//   lane = 16*((k>>3)&1) + (col&15), elem = (k&7) | ((k>>4)<<3), ntile = col>>4.
// Each lane then reads its whole fragment as one contiguous 32B chunk.

// C[M,N] = A[M,K] (f32, lda) x B[K,N] (f32, ldb) + bias[N] (+ resid[M,N]).
__global__ __launch_bounds__(256) void ve_gemm(
    const float* __restrict__ A, int lda,
    const float* __restrict__ Bw, int ldb,
    const float* __restrict__ bias,
    const float* __restrict__ resid,
    float* __restrict__ C, int ldc,
    int M, int N, int K) {
    __shared__ _Float16 Af[4][32][16];   // [msub][lane][elem]  4 KB
    __shared__ _Float16 Bf[8][32][16];   // [ntile][lane][elem] 8 KB
    const int tid  = threadIdx.x;
    const int lane = tid & 31;
    const int wave = tid >> 5;
    const int msub = wave & 3;           // 16-row strip
    const int nsub = wave >> 2;          // 64-col strip
    const int tm = blockIdx.x * TILE_M;
    const int tn = blockIdx.y * TILE_N;

    v8f acc[4];
    #pragma unroll
    for (int nt = 0; nt < 4; ++nt)
        #pragma unroll
        for (int r = 0; r < 8; ++r) acc[nt][r] = 0.f;

    for (int k0 = 0; k0 < K; k0 += TILE_K) {
        // stage A: 64x32, float4 loads, v4h fragment-order stores (2 iters/thr)
        for (int i = tid; i < (TILE_M * TILE_K) / 4; i += 256) {
            int r  = i >> 3;             // row in tile 0..63
            int cg = (i & 7) << 2;       // k group 0,4,...,28
            int gr = tm + r;
            float4 v = make_float4(0.f, 0.f, 0.f, 0.f);
            if (gr < M) {
                v = *(const float4*)&A[(size_t)gr * lda + k0 + cg];
                if (k0 + TILE_K < K)
                    __builtin_prefetch(&A[(size_t)gr * lda + k0 + TILE_K + cg], 0, 1);
            }
            int ln = (((cg >> 3) & 1) << 4) + (r & 15);
            int e0 = (cg & 7) | ((cg >> 4) << 3);
            v4h hv;
            hv[0] = (_Float16)v.x; hv[1] = (_Float16)v.y;
            hv[2] = (_Float16)v.z; hv[3] = (_Float16)v.w;
            *(v4h*)&Af[r >> 4][ln][e0] = hv;
        }
        // stage B: 32x128, float4 loads along N (coalesced), scatter u16 (4/thr)
        for (int i = tid; i < (TILE_K * TILE_N) / 4; i += 256) {
            int r  = i >> 5;             // k row 0..31
            int cg = (i & 31) << 2;      // col group 0,4,...,124
            float4 v = *(const float4*)&Bw[(size_t)(k0 + r) * ldb + tn + cg];
            if (k0 + TILE_K < K)
                __builtin_prefetch(&Bw[(size_t)(k0 + TILE_K + r) * ldb + tn + cg], 0, 1);
            int lh = ((r >> 3) & 1) << 4;
            int e  = (r & 7) | ((r >> 4) << 3);
            int nt = cg >> 4;            // same tile for all 4 (cg % 16 in {0,4,8,12})
            Bf[nt][lh + ((cg + 0) & 15)][e] = (_Float16)v.x;
            Bf[nt][lh + ((cg + 1) & 15)][e] = (_Float16)v.y;
            Bf[nt][lh + ((cg + 2) & 15)][e] = (_Float16)v.z;
            Bf[nt][lh + ((cg + 3) & 15)][e] = (_Float16)v.w;
        }
        __syncthreads();

        const v16h af = *(const v16h*)&Af[msub][lane][0];
        #pragma unroll
        for (int nt = 0; nt < 4; ++nt) {
            const v16h bf = *(const v16h*)&Bf[nsub * 4 + nt][lane][0];
            acc[nt] = __builtin_amdgcn_wmma_f32_16x16x32_f16(
                false, af, false, bf, (short)0, acc[nt], false, false);
        }
        __syncthreads();
    }

    #pragma unroll
    for (int nt = 0; nt < 4; ++nt) {
        int col = tn + nsub * 64 + nt * 16 + (lane & 15);
        float bv = bias ? bias[col] : 0.f;
        #pragma unroll
        for (int r = 0; r < 8; ++r) {
            int row = tm + msub * 16 + (lane >> 4) * 8 + r;
            if (row < M) {
                float v = acc[nt][r] + bv;
                if (resid) v += resid[(size_t)row * ldc + col];
                C[(size_t)row * ldc + col] = v;
            }
        }
    }
}

// Patch-embed GEMM: A = im2col of f2 [8192 x 4096] gathered on the fly,
// B = patch_w^T (patch_w is [512, 4096] = [N, K] so K is contiguous -> float4).
__global__ __launch_bounds__(256) void ve_patch_gemm(
    const float* __restrict__ f2,   // [32,64,128,128]
    const float* __restrict__ pw,   // [512,64,8,8]
    const float* __restrict__ pb,   // [512]
    float* __restrict__ pe) {       // [8192, 512]
    __shared__ _Float16 Af[4][32][16];
    __shared__ _Float16 Bf[8][32][16];
    const int tid  = threadIdx.x;
    const int lane = tid & 31;
    const int wave = tid >> 5;
    const int msub = wave & 3;
    const int nsub = wave >> 2;
    const int tm = blockIdx.x * TILE_M;
    const int tn = blockIdx.y * TILE_N;
    const int K = 4096;

    v8f acc[4];
    #pragma unroll
    for (int nt = 0; nt < 4; ++nt)
        #pragma unroll
        for (int r = 0; r < 8; ++r) acc[nt][r] = 0.f;

    for (int k0 = 0; k0 < K; k0 += TILE_K) {
        // A: scattered im2col gather, single u16 store into fragment slot
        for (int i = tid; i < TILE_M * TILE_K; i += 256) {
            int r = i >> 5, c = i & 31;
            int row = tm + r;
            int k = k0 + c;
            int bn = row >> 8;
            int t  = row & 255;
            int gy = t >> 4, gx = t & 15;
            int ch = k >> 6;
            int rr = k & 63;
            int py = rr >> 3, px = rr & 7;
            float v = f2[(((size_t)(bn * 64 + ch) * 128) + gy * 8 + py) * 128
                         + gx * 8 + px];
            Af[r >> 4][(((c >> 3) & 1) << 4) + (r & 15)][(c & 7) | ((c >> 4) << 3)]
                = (_Float16)v;
        }
        // B: float4 along K (contiguous in pw), v4h fragment-order store
        for (int i = tid; i < (TILE_K * TILE_N) / 4; i += 256) {
            int c  = i >> 3;             // col 0..127
            int kg = (i & 7) << 2;       // k group 0,4,...,28
            float4 v = *(const float4*)&pw[(size_t)(tn + c) * 4096 + k0 + kg];
            v4h hv;
            hv[0] = (_Float16)v.x; hv[1] = (_Float16)v.y;
            hv[2] = (_Float16)v.z; hv[3] = (_Float16)v.w;
            *(v4h*)&Bf[c >> 4][(((kg >> 3) & 1) << 4) + (c & 15)][(kg & 7) | ((kg >> 4) << 3)] = hv;
        }
        __syncthreads();

        const v16h af = *(const v16h*)&Af[msub][lane][0];
        #pragma unroll
        for (int nt = 0; nt < 4; ++nt) {
            const v16h bf = *(const v16h*)&Bf[nsub * 4 + nt][lane][0];
            acc[nt] = __builtin_amdgcn_wmma_f32_16x16x32_f16(
                false, af, false, bf, (short)0, acc[nt], false, false);
        }
        __syncthreads();
    }
    #pragma unroll
    for (int nt = 0; nt < 4; ++nt) {
        int col = tn + nsub * 64 + nt * 16 + (lane & 15);
        float bv = pb[col];
        #pragma unroll
        for (int r = 0; r < 8; ++r) {
            int row = tm + msub * 16 + (lane >> 4) * 8 + r;
            pe[(size_t)row * 512 + col] = acc[nt][r] + bv;
        }
    }
}

// ----------------------- token assembly (cls + pos2d) ----------------------
__global__ void ve_assemble_xa(const float* __restrict__ pe, const float* __restrict__ cls,
                               float* __restrict__ xa) {
    int idx = blockIdx.x * blockDim.x + threadIdx.x;
    int total = BATCH * NTOK * DMODEL;
    if (idx >= total) return;
    int d = idx % DMODEL;
    int t = (idx / DMODEL) % NTOK;
    int bn = idx / (DMODEL * NTOK);
    if (t == 0) { xa[idx] = cls[d]; return; }
    int tt = t - 1;
    float yv = (float)(tt % 16);   // reference: y = tile(arange)
    float xv = (float)(tt / 16);   // reference: x = repeat(arange)
    int q = d / 128;
    int j = d % 128;
    float omega = expf(-logf(10000.f) * (float)j / 128.f);
    float pos;
    if      (q == 0) pos = sinf(xv * omega);
    else if (q == 1) pos = cosf(xv * omega);
    else if (q == 2) pos = sinf(yv * omega);
    else             pos = cosf(yv * omega);
    xa[idx] = pe[(size_t)(bn * 256 + tt) * 512 + d] + pos;
}

// ----------------------------- LayerNorm -----------------------------------
__global__ __launch_bounds__(256) void ve_layernorm(const float* __restrict__ x,
                                                    const float* __restrict__ g,
                                                    const float* __restrict__ b,
                                                    float* __restrict__ o) {
    int row = blockIdx.x;
    const float* xr = x + (size_t)row * DMODEL;
    float s = 0.f, ss = 0.f;
    for (int i = threadIdx.x; i < DMODEL; i += 256) {
        float v = xr[i]; s += v; ss += v * v;
    }
    block_sum2(s, ss);
    float m = s / (float)DMODEL;
    float inv = rsqrtf(ss / (float)DMODEL - m * m + EPSV);
    for (int i = threadIdx.x; i < DMODEL; i += 256)
        o[(size_t)row * DMODEL + i] = (xr[i] - m) * inv * g[i] + b[i];
}

// ----------------------------- xPos rope -----------------------------------
// Matches reference quirk: tables are length n_pos = H, indexed by head id.
__global__ void ve_rope_xpos(float* __restrict__ qkv) {
    int idx = blockIdx.x * blockDim.x + threadIdx.x;
    int total = BATCH * NTOK * NHEAD * (HDIM / 2);
    if (idx >= total) return;
    int j = idx & 31;               // pair index
    int h = (idx >> 5) & 7;
    int bt = idx >> 8;              // bn*257 + t
    float invf = expf(-logf(10000.f) * (2.f * (float)j) / 64.f);
    float ang = (float)h * invf;
    float cv = cosf(ang), snv = sinf(ang);
    float p = ((float)h - 4.f) / 512.f;
    int c0 = 2 * j, c1 = 2 * j + 1;
    float sc0 = powf((2.f * (float)(c0 & 31) + 25.6f) / 89.6f, p);
    float sc1 = powf((2.f * (float)(c1 & 31) + 25.6f) / 89.6f, p);
    size_t base = (size_t)bt * 1536 + (size_t)h * 64;
    float q0 = qkv[base + c0], q1 = qkv[base + c1];
    qkv[base + c0] = (q0 * cv - q1 * snv) * sc0;
    qkv[base + c1] = (q1 * cv + q0 * snv) * sc1;
    size_t kb = base + 512;
    float k0 = qkv[kb + c0], k1 = qkv[kb + c1];
    qkv[kb + c0] = (k0 * cv - k1 * snv) / sc0;
    qkv[kb + c1] = (k1 * cv + k0 * snv) / sc1;
}

// ----------------------------- attention -----------------------------------
__global__ __launch_bounds__(256) void ve_attention(const float* __restrict__ qkv,
                                                    float* __restrict__ y) {
    int id = blockIdx.x;
    int i  = id % NTOK;
    int h  = (id / NTOK) % NHEAD;
    int bn = id / (NTOK * NHEAD);
    int tid = threadIdx.x;
    __shared__ float qv[HDIM];
    __shared__ float pr[NTOK];
    const size_t tb = (size_t)bn * NTOK * 1536;
    if (tid < HDIM) qv[tid] = qkv[tb + (size_t)i * 1536 + h * 64 + tid];
    __syncthreads();
    float lmax = -3.4e38f;
    for (int j = tid; j < NTOK; j += 256) {
        const float4* kr4 = (const float4*)(qkv + tb + (size_t)j * 1536 + 512 + h * 64);
        float s = 0.f;
        #pragma unroll
        for (int d = 0; d < HDIM / 4; ++d) {
            float4 kv = kr4[d];
            s += qv[4 * d + 0] * kv.x + qv[4 * d + 1] * kv.y
               + qv[4 * d + 2] * kv.z + qv[4 * d + 3] * kv.w;
        }
        s *= 0.125f;                  // HD^-0.5
        pr[j] = s;
        lmax = fmaxf(lmax, s);
    }
    float mx = block_max1(lmax);
    float lsum = 0.f;
    for (int j = tid; j < NTOK; j += 256) {
        float e = expf(pr[j] - mx);
        pr[j] = e;
        lsum += e;
    }
    float sm = block_sum1(lsum);
    if (tid < HDIM) {
        float acc = 0.f;
        const float* vb = qkv + tb + 1024 + h * 64 + tid;
        for (int j = 0; j < NTOK; ++j) acc += pr[j] * vb[(size_t)j * 1536];
        y[((size_t)bn * NTOK + i) * DMODEL + h * 64 + tid] = acc / sm;
    }
}

// ----------------------------- SwiGLU --------------------------------------
__global__ void ve_swiglu(const float* __restrict__ h1, const float* __restrict__ h2,
                          float* __restrict__ hm, int n) {
    int idx = blockIdx.x * blockDim.x + threadIdx.x;
    if (idx >= n) return;
    float v = h1[idx];
    hm[idx] = (v / (1.f + expf(-v))) * h2[idx];
}

// --------------------- final LN + (cls, tokens) split ----------------------
__global__ __launch_bounds__(256) void ve_final_ln_out(const float* __restrict__ xa,
                                                       const float* __restrict__ g,
                                                       const float* __restrict__ b,
                                                       float* __restrict__ out) {
    int row = blockIdx.x;
    int bn = row / NTOK, t = row % NTOK;
    const float* xr = xa + (size_t)row * DMODEL;
    float s = 0.f, ss = 0.f;
    for (int i = threadIdx.x; i < DMODEL; i += 256) {
        float v = xr[i]; s += v; ss += v * v;
    }
    block_sum2(s, ss);
    float m = s / (float)DMODEL;
    float inv = rsqrtf(ss / (float)DMODEL - m * m + EPSV);
    float* dst = (t == 0) ? (out + (size_t)bn * DMODEL)
                          : (out + (size_t)BATCH * DMODEL
                                 + ((size_t)bn * 256 + (t - 1)) * DMODEL);
    for (int i = threadIdx.x; i < DMODEL; i += 256)
        dst[i] = (xr[i] - m) * inv * g[i] + b[i];
}

// ============================================================================
extern "C" void kernel_launch(void* const* d_in, const int* in_sizes, int n_in,
                              void* d_out, int out_size, void* d_ws, size_t ws_size,
                              hipStream_t stream) {
    (void)in_sizes; (void)n_in; (void)out_size; (void)ws_size;
    const float* x       = (const float*)d_in[0];
    const float* conv1_w = (const float*)d_in[1];
    const float* conv1_b = (const float*)d_in[2];
    const float* gn1_g   = (const float*)d_in[3];
    const float* gn1_b   = (const float*)d_in[4];
    const float* conv2_w = (const float*)d_in[5];
    const float* conv2_b = (const float*)d_in[6];
    const float* gn2_g   = (const float*)d_in[7];
    const float* gn2_b   = (const float*)d_in[8];
    const float* patch_w = (const float*)d_in[9];
    const float* patch_b = (const float*)d_in[10];
    const float* cls_tok = (const float*)d_in[11];
    const float* ln1_g   = (const float*)d_in[12];
    const float* ln1_b   = (const float*)d_in[13];
    const float* qkv_w   = (const float*)d_in[14];
    const float* qkv_b   = (const float*)d_in[15];
    const float* out_w   = (const float*)d_in[16];
    const float* out_b   = (const float*)d_in[17];
    const float* ln2_g   = (const float*)d_in[18];
    const float* ln2_b   = (const float*)d_in[19];
    const float* w1_w    = (const float*)d_in[20];
    const float* w1_b    = (const float*)d_in[21];
    const float* w2_w    = (const float*)d_in[22];
    const float* w2_b    = (const float*)d_in[23];
    const float* w3_w    = (const float*)d_in[24];
    const float* w3_b    = (const float*)d_in[25];
    const float* fn_g    = (const float*)d_in[26];
    const float* fn_b    = (const float*)d_in[27];

    // Workspace layout (floats). Pool of 3 MLP slots, reused by the conv stem.
    const size_t H_ELEMS = (size_t)MROWS * MLPD;        // 16,842,752
    float* ws  = (float*)d_ws;
    float* f1  = ws;                                    // conv1 out (16.78M)
    float* f2  = ws + H_ELEMS;                          // conv2 out (33.55M <= 2 slots)
    float* h1  = ws;                                    // MLP h1 (aliases f1, dead)
    float* h2  = ws + H_ELEMS;                          // MLP h2 (aliases f2, dead)
    float* hm  = ws + 2 * H_ELEMS;                      // MLP gated
    float* xa  = ws + 3 * H_ELEMS;                      // residual stream [8224,512]
    float* nb  = xa  + (size_t)MROWS * DMODEL;          // LN scratch
    float* qkv = nb  + (size_t)MROWS * DMODEL;          // [8224,1536]
    float* pe  = qkv;                                   // patch-embed out (reuse)
    float* yb  = qkv + (size_t)MROWS * 3 * DMODEL;      // attn out [8224,512]
    float* st  = yb  + (size_t)MROWS * DMODEL;          // GN stats

    // --- conv stem ---
    {
        int total = BATCH * 32 * 128 * 128;
        ve_conv3x3<<<(total + 255) / 256, 256, 0, stream>>>(x, conv1_w, conv1_b, f1, 3, 32);
        ve_gn_stats<<<BATCH * 4, 256, 0, stream>>>(f1, st, 32, 4);
        ve_gn_apply_gelu<<<(total + 255) / 256, 256, 0, stream>>>(f1, st, gn1_g, gn1_b, 32, 4);
    }
    {
        int total = BATCH * 64 * 128 * 128;
        ve_conv3x3<<<(total + 255) / 256, 256, 0, stream>>>(f1, conv2_w, conv2_b, f2, 32, 64);
        ve_gn_stats<<<BATCH * 8, 256, 0, stream>>>(f2, st, 64, 8);
        ve_gn_apply_gelu<<<(total + 255) / 256, 256, 0, stream>>>(f2, st, gn2_g, gn2_b, 64, 8);
    }

    // --- patch embed (WMMA GEMM 8192x512x4096) + token assembly ---
    {
        dim3 grid(8192 / TILE_M, 512 / TILE_N);
        ve_patch_gemm<<<grid, 256, 0, stream>>>(f2, patch_w, patch_b, pe);
        int total = BATCH * NTOK * DMODEL;
        ve_assemble_xa<<<(total + 255) / 256, 256, 0, stream>>>(pe, cls_tok, xa);
    }

    // --- transformer layers ---
    const int M = MROWS;
    const int gM = (M + TILE_M - 1) / TILE_M;   // 129
    for (int l = 0; l < NLAYER; ++l) {
        ve_layernorm<<<M, 256, 0, stream>>>(xa, ln1_g + l * DMODEL, ln1_b + l * DMODEL, nb);

        dim3 gq(gM, (3 * DMODEL) / TILE_N);
        ve_gemm<<<gq, 256, 0, stream>>>(nb, DMODEL,
                                        qkv_w + (size_t)l * DMODEL * 3 * DMODEL, 3 * DMODEL,
                                        qkv_b + (size_t)l * 3 * DMODEL, nullptr,
                                        qkv, 3 * DMODEL, M, 3 * DMODEL, DMODEL);

        int rt = BATCH * NTOK * NHEAD * (HDIM / 2);
        ve_rope_xpos<<<(rt + 255) / 256, 256, 0, stream>>>(qkv);

        ve_attention<<<BATCH * NHEAD * NTOK, 256, 0, stream>>>(qkv, yb);

        dim3 go(gM, DMODEL / TILE_N);
        ve_gemm<<<go, 256, 0, stream>>>(yb, DMODEL,
                                        out_w + (size_t)l * DMODEL * DMODEL, DMODEL,
                                        out_b + (size_t)l * DMODEL, xa,
                                        xa, DMODEL, M, DMODEL, DMODEL);

        ve_layernorm<<<M, 256, 0, stream>>>(xa, ln2_g + l * DMODEL, ln2_b + l * DMODEL, nb);

        dim3 gm(gM, MLPD / TILE_N);
        ve_gemm<<<gm, 256, 0, stream>>>(nb, DMODEL,
                                        w1_w + (size_t)l * DMODEL * MLPD, MLPD,
                                        w1_b + (size_t)l * MLPD, nullptr,
                                        h1, MLPD, M, MLPD, DMODEL);
        ve_gemm<<<gm, 256, 0, stream>>>(nb, DMODEL,
                                        w2_w + (size_t)l * DMODEL * MLPD, MLPD,
                                        w2_b + (size_t)l * MLPD, nullptr,
                                        h2, MLPD, M, MLPD, DMODEL);

        int nsw = M * MLPD;
        ve_swiglu<<<(nsw + 255) / 256, 256, 0, stream>>>(h1, h2, hm, nsw);

        ve_gemm<<<go, 256, 0, stream>>>(hm, MLPD,
                                        w3_w + (size_t)l * MLPD * DMODEL, DMODEL,
                                        w3_b + (size_t)l * DMODEL, xa,
                                        xa, DMODEL, M, DMODEL, MLPD);
    }

    ve_final_ln_out<<<M, 256, 0, stream>>>(xa, fn_g, fn_b, (float*)d_out);
}
